// LatentAttentionBlock_37632503447610
// MI455X (gfx1250) — matrix operand order repaired and checked by reading, compiled, not run on hardware
//
#include <hip/hip_runtime.h>
#include <hip/hip_bf16.h>
#include <math.h>

typedef __attribute__((ext_vector_type(16))) __bf16 v16bf;
typedef __attribute__((ext_vector_type(8)))  float  v8f;

#define DIM   1024
#define QK    128
#define EXPD  2048
#define EDIM  4352        // 2*QK + 2*EXPD
#define SEQ   4096
#define SEQP  4224        // SEQ + 128 pad (tail chunk safety for PV loads)
#define BATCH 4
#define ROWS  (BATCH*SEQ) // 16384

union FragU { v16bf v; uint4 u[2]; };

// A-matrix 16x32 bf16 fragment (ISA 7.12.2): lane m<16 holds K=[0..7]+[16..23],
// lane m+16 holds K=[8..15]+[24..31]; rows are K-contiguous in memory.
__device__ __forceinline__ v16bf load_a_frag(const __bf16* A, int ldk, int row0, int k0, int lane) {
  int r = lane & 15, hh = (lane >> 4) & 1;
  const __bf16* p = A + (size_t)(row0 + r) * ldk + (k0 + hh * 8);
  FragU f;
  f.u[0] = *(const uint4*)(p);
  f.u[1] = *(const uint4*)(p + 16);
  return f.v;
}

// B-matrix 32x16 bf16 fragment: lane n<16 holds K=[0..7]+[16..23] of column n, lane n+16
// holds K=[8..15]+[24..31]. Source is stored [N][K] row-major (K contiguous) => 2x B128.
__device__ __forceinline__ v16bf load_b_frag(const __bf16* B, int ldk, int col0, int k0, int lane) {
  int n = lane & 15, hh = (lane >> 4) & 1;
  const __bf16* p = B + (size_t)(col0 + n) * ldk + (k0 + hh * 8);
  FragU f;
  f.u[0] = *(const uint4*)(p);
  f.u[1] = *(const uint4*)(p + 16);
  return f.v;
}

__device__ __forceinline__ v8f wmma_bf16(v16bf a, v16bf b, v8f c) {
  return __builtin_amdgcn_wmma_f32_16x16x32_bf16(false, a, false, b, (short)0, c, false, false);
}

// ---------------------------------------------------------------- cast / fill
__global__ void __launch_bounds__(256) cast_bf16_kernel(const float* __restrict__ in,
                                                        __bf16* __restrict__ out, int n) {
  int i = blockIdx.x * blockDim.x + threadIdx.x;
  if (i < n) out[i] = (__bf16)in[i];
}

__global__ void __launch_bounds__(256) zero_bf16_kernel(__bf16* __restrict__ p, int n) {
  int i = blockIdx.x * blockDim.x + threadIdx.x;
  if (i < n) p[i] = (__bf16)0.0f;
}

// ---------------------------------------------------------------- layernorm
__global__ void __launch_bounds__(256) ln_kernel(const float* __restrict__ x,
                                                 const float* __restrict__ gamma,
                                                 const float* __restrict__ beta,
                                                 __bf16* __restrict__ h) {
  __shared__ float red[32];
  int row = blockIdx.x;
  const float* xr = x + (size_t)row * DIM;
  int t = threadIdx.x;
  float v[4];
  float s = 0.f, ss = 0.f;
#pragma unroll
  for (int j = 0; j < 4; ++j) {
    v[j] = xr[t + j * 256];
    s += v[j];
    ss += v[j] * v[j];
  }
#pragma unroll
  for (int m = 1; m < 32; m <<= 1) {
    s  += __shfl_xor(s,  m, 32);
    ss += __shfl_xor(ss, m, 32);
  }
  int lane = t & 31, wid = t >> 5;
  if (lane == 0) { red[wid] = s; red[8 + wid] = ss; }
  __syncthreads();
  if (t == 0) {
    float S = 0.f, SS = 0.f;
    for (int w = 0; w < 8; ++w) { S += red[w]; SS += red[8 + w]; }
    float mu = S / (float)DIM;
    red[16] = mu;
    red[17] = rsqrtf(SS / (float)DIM - mu * mu + 1e-5f);
  }
  __syncthreads();
  float mu = red[16], rstd = red[17];
#pragma unroll
  for (int j = 0; j < 4; ++j) {
    int col = t + j * 256;
    h[(size_t)row * DIM + col] = (__bf16)((v[j] - mu) * rstd * gamma[col] + beta[col]);
  }
}

// ------------------------------------------------------- generic WMMA GEMM
// C[M,N](f32) = A[M,K](bf16,row) * B[N,K](bf16,row)^T  (+ compile-time residual)
// block: 256 thr = 8 waves (2M x 4N), wave tile 64x64, block tile 128x256.
template <bool HAS_RESID>
__global__ void __launch_bounds__(256, 1)
wmma_gemm_kernel(const __bf16* __restrict__ A, const __bf16* __restrict__ B,
                 float* __restrict__ C, const float* __restrict__ resid,
                 int N, int K) {
  int lane = threadIdx.x & 31;
  int wid  = threadIdx.x >> 5;
  int wm = wid >> 2, wn = wid & 3;
  int row0 = blockIdx.y * 128 + wm * 64;
  int col0 = blockIdx.x * 256 + wn * 64;
  v8f acc[4][4] = {};
  for (int k0 = 0; k0 < K; k0 += 32) {
    v16bf a[4], b[4];
#pragma unroll
    for (int i = 0; i < 4; ++i) a[i] = load_a_frag(A, K, row0 + i * 16, k0, lane);
#pragma unroll
    for (int j = 0; j < 4; ++j) b[j] = load_b_frag(B, K, col0 + j * 16, k0, lane);
#pragma unroll
    for (int i = 0; i < 4; ++i)
#pragma unroll
      for (int j = 0; j < 4; ++j)
        acc[i][j] = wmma_bf16(a[i], b[j], acc[i][j]);
  }
  int c = lane & 15, hh = (lane >> 4) & 1;
#pragma unroll
  for (int i = 0; i < 4; ++i) {
#pragma unroll
    for (int j = 0; j < 4; ++j) {
      size_t base = (size_t)(row0 + i * 16 + hh * 8) * N + (size_t)(col0 + j * 16 + c);
      float* __restrict__ cp = C + base;
      const float* __restrict__ rp = HAS_RESID ? (resid + base) : nullptr;
#pragma unroll
      for (int e = 0; e < 8; ++e) {
        float val = acc[i][j][e];
        if (HAS_RESID) val += rp[0];
        cp[0] = val;
        cp += N;
        if (HAS_RESID) rp += N;
      }
    }
  }
}

// ---------------------------------------------------------------- GeGLU split
__global__ void __launch_bounds__(256) geglu_kernel(const float* __restrict__ proj,
                                                    __bf16* __restrict__ qb, __bf16* __restrict__ kb,
                                                    __bf16* __restrict__ concat,
                                                    __bf16* __restrict__ vt) {
  int row = blockIdx.x;            // 0..ROWS-1
  int b = row >> 12, s = row & (SEQ - 1);
  const float* pr = proj + (size_t)row * EDIM;
  int t = threadIdx.x;
  if (t < QK) {
    qb[(size_t)row * QK + t] = (__bf16)pr[t];
    kb[(size_t)row * QK + t] = (__bf16)pr[QK + t];
  }
  for (int e = t; e < EXPD; e += 256) {
    float lin = pr[2 * QK + e];
    float pre = pr[2 * QK + EXPD + e];
    float g = lin * 0.5f * pre * (1.0f + erff(pre * 0.70710678118654752f));
    if (e < EXPD - DIM) {
      concat[(size_t)row * (2 * DIM) + e] = (__bf16)g;          // geglu_local
    } else {
      int vc = e - (EXPD - DIM);                                 // 0..DIM-1
      vt[((size_t)b * DIM + vc) * SEQP + s] = (__bf16)g;         // V transposed
    }
  }
}

// --------------------------------------------------- attention pass A: stats
__global__ void __launch_bounds__(256, 1)
attn_stats_kernel(const __bf16* __restrict__ qb, const __bf16* __restrict__ kb,
                  const float* __restrict__ pbm,
                  float* __restrict__ sm, float* __restrict__ sl) {
  int lane = threadIdx.x & 31;
  int wid  = threadIdx.x >> 5;
  int task = blockIdx.x * 8 + wid;     // 0..1023
  int b  = task >> 8;
  int qt = (task & 255) * 16;
  int qbase = b * SEQ + qt;
  const float scale = 0.08838834764831845f;   // 1/sqrt(128)
  float sp = log1pf(__expf(pbm[0]));
  v16bf qf[4];
#pragma unroll
  for (int kk = 0; kk < 4; ++kk) qf[kk] = load_a_frag(qb, QK, qbase, kk * 32, lane);
  int c = lane & 15, hh = (lane >> 4) & 1;
  float m[8], l[8];
#pragma unroll
  for (int i = 0; i < 8; ++i) { m[i] = -1e30f; l[i] = 0.f; }
  for (int j0 = 0; j0 <= qt; j0 += 16) {
    v8f acc = {};
#pragma unroll
    for (int kk = 0; kk < 4; ++kk) {
      v16bf kf = load_b_frag(kb, QK, b * SEQ + j0, kk * 32, lane);
      acc = wmma_bf16(qf[kk], kf, acc);
    }
    int j = j0 + c;
#pragma unroll
    for (int i = 0; i < 8; ++i) {
      int iq = qt + hh * 8 + i;
      bool ok = (j <= iq);
      float sc = acc[i] * scale + sp * (float)(j - iq);
      float val = ok ? sc : -1e30f;
      float tmax = val;
      tmax = fmaxf(tmax, __shfl_xor(tmax, 1, 32));
      tmax = fmaxf(tmax, __shfl_xor(tmax, 2, 32));
      tmax = fmaxf(tmax, __shfl_xor(tmax, 4, 32));
      tmax = fmaxf(tmax, __shfl_xor(tmax, 8, 32));
      float nm = fmaxf(m[i], tmax);
      float p = ok ? __expf(sc - nm) : 0.f;
      p += __shfl_xor(p, 1, 32);
      p += __shfl_xor(p, 2, 32);
      p += __shfl_xor(p, 4, 32);
      p += __shfl_xor(p, 8, 32);
      l[i] = l[i] * __expf(m[i] - nm) + p;
      m[i] = nm;
    }
  }
  if (c == 0) {
#pragma unroll
    for (int i = 0; i < 8; ++i) {
      int r = qbase + hh * 8 + i;
      sm[r] = m[i];
      sl[r] = l[i];
    }
  }
}

// ------------------------------------------------------ attention pass B: PV
// block = 16 query rows, 8 waves. Each wave builds one 16-key slice of the
// normalized P chunk (LDS), then all waves consume P against their 128-wide
// V^T slice. Output written into concat columns [DIM, 2*DIM).
__global__ void __launch_bounds__(256, 1)
attn_pv_kernel(const __bf16* __restrict__ qb, const __bf16* __restrict__ kb,
               const __bf16* __restrict__ vt, const float* __restrict__ pbm,
               const float* __restrict__ sm, const float* __restrict__ sl,
               __bf16* __restrict__ concat) {
  __shared__ __align__(16) __bf16 Pch[16 * 128];
  int lane = threadIdx.x & 31;
  int wid  = threadIdx.x >> 5;
  int task = blockIdx.x;               // 0..1023
  int b  = task >> 8;
  int qt = (task & 255) * 16;
  int qbase = b * SEQ + qt;
  const float scale = 0.08838834764831845f;
  float sp = log1pf(__expf(pbm[0]));
  int c = lane & 15, hh = (lane >> 4) & 1;
  float mrow[8], linv[8];
#pragma unroll
  for (int i = 0; i < 8; ++i) {
    int r = qbase + hh * 8 + i;
    mrow[i] = sm[r];
    linv[i] = 1.0f / sl[r];
  }
  v16bf qf[4];
#pragma unroll
  for (int kk = 0; kk < 4; ++kk) qf[kk] = load_a_frag(qb, QK, qbase, kk * 32, lane);
  int vcol0 = wid * 128;
  const __bf16* vb = vt + (size_t)b * DIM * SEQP;
  v8f o[8] = {};
  for (int j0 = 0; j0 <= qt; j0 += 128) {
    int jk = j0 + wid * 16;
    if (jk <= qt) {
      v8f acc = {};
#pragma unroll
      for (int kk = 0; kk < 4; ++kk) {
        v16bf kf = load_b_frag(kb, QK, b * SEQ + jk, kk * 32, lane);
        acc = wmma_bf16(qf[kk], kf, acc);
      }
      int j = jk + c;
#pragma unroll
      for (int i = 0; i < 8; ++i) {
        int iq = qt + hh * 8 + i;
        bool ok = (j <= iq);
        float sc = acc[i] * scale + sp * (float)(j - iq);
        float p = ok ? __expf(sc - mrow[i]) * linv[i] : 0.f;
        Pch[(hh * 8 + i) * 128 + wid * 16 + c] = (__bf16)p;
      }
    } else {
#pragma unroll
      for (int i = 0; i < 8; ++i)
        Pch[(hh * 8 + i) * 128 + wid * 16 + c] = (__bf16)0.0f;
    }
    __syncthreads();
#pragma unroll
    for (int kk = 0; kk < 4; ++kk) {
      v16bf pf = load_a_frag(Pch, 128, 0, kk * 32, lane);
#pragma unroll
      for (int nt = 0; nt < 8; ++nt) {
        v16bf vf = load_b_frag(vb, SEQP, vcol0 + nt * 16, j0 + kk * 32, lane);
        o[nt] = wmma_bf16(pf, vf, o[nt]);
      }
    }
    __syncthreads();
  }
#pragma unroll
  for (int nt = 0; nt < 8; ++nt) {
#pragma unroll
    for (int i = 0; i < 8; ++i) {
      size_t idx = (size_t)(qbase + hh * 8 + i) * (2 * DIM) + (size_t)(DIM + vcol0 + nt * 16 + c);
      concat[idx] = (__bf16)o[nt][i];
    }
  }
}

// ---------------------------------------------------------------- launcher
extern "C" void kernel_launch(void* const* d_in, const int* in_sizes, int n_in,
                              void* d_out, int out_size, void* d_ws, size_t ws_size,
                              hipStream_t stream) {
  (void)in_sizes; (void)n_in; (void)out_size; (void)ws_size;
  const float* x       = (const float*)d_in[0];
  const float* expand  = (const float*)d_in[1];
  const float* project = (const float*)d_in[2];
  const float* gamma   = (const float*)d_in[3];
  const float* beta    = (const float*)d_in[4];
  const float* pbm     = (const float*)d_in[5];
  float* out = (float*)d_out;

  char* ws = (char*)d_ws;
  size_t off = 0;
  auto alloc = [&](size_t bytes) -> void* {
    void* p = ws + off;
    off += (bytes + 255) & ~(size_t)255;
    return p;
  };
  __bf16* h      = (__bf16*)alloc((size_t)ROWS * DIM * 2);
  __bf16* exb    = (__bf16*)alloc((size_t)EDIM * DIM * 2);
  __bf16* prb    = (__bf16*)alloc((size_t)DIM * EXPD * 2);
  float*  proj   = (float*) alloc((size_t)ROWS * EDIM * 4);
  __bf16* qb     = (__bf16*)alloc((size_t)ROWS * QK * 2);
  __bf16* kb     = (__bf16*)alloc((size_t)ROWS * QK * 2);
  __bf16* concat = (__bf16*)alloc((size_t)ROWS * 2 * DIM * 2);
  __bf16* vt     = (__bf16*)alloc((size_t)BATCH * DIM * SEQP * 2);
  float*  sm     = (float*) alloc((size_t)ROWS * 4);
  float*  sl     = (float*) alloc((size_t)ROWS * 4);

  int nE = EDIM * DIM, nP = DIM * EXPD, nVt = BATCH * DIM * SEQP;
  cast_bf16_kernel<<<(nE + 255) / 256, 256, 0, stream>>>(expand, exb, nE);
  cast_bf16_kernel<<<(nP + 255) / 256, 256, 0, stream>>>(project, prb, nP);
  zero_bf16_kernel<<<(nVt + 255) / 256, 256, 0, stream>>>(vt, nVt);
  ln_kernel<<<ROWS, 256, 0, stream>>>(x, gamma, beta, h);
  wmma_gemm_kernel<false><<<dim3(EDIM / 256, ROWS / 128), 256, 0, stream>>>(h, exb, proj, nullptr, EDIM, DIM);
  geglu_kernel<<<ROWS, 256, 0, stream>>>(proj, qb, kb, concat, vt);
  attn_stats_kernel<<<128, 256, 0, stream>>>(qb, kb, pbm, sm, sl);
  attn_pv_kernel<<<1024, 256, 0, stream>>>(qb, kb, vt, pbm, sm, sl, concat);
  wmma_gemm_kernel<true><<<dim3(DIM / 256, ROWS / 128), 256, 0, stream>>>(concat, prb, out, x, DIM, 2 * DIM);
}
